// PointerGeneratorNetwork_31559419691416
// MI455X (gfx1250) — matrix operand order repaired and checked by reading, not compiled, not run on hardware
//
#include <hip/hip_runtime.h>
#include <hip/hip_bf16.h>

// ---------------------------------------------------------------------------
// Pointer-Generator Network for MI455X (gfx1250, wave32, WMMA).
// fp16 weights/activations -> v_wmma_f32_16x16x32_f16, fp32 accumulation.
// Weights are L2-resident (out2_W fp16 = 25.6MB << 192MB L2), so per-step
// weight re-loads are cheap; we deliberately prevent LICM from hoisting the
// weight fragments out of the recurrent loops (register spill to scratch).
// Async global->LDS copies (ASYNCcnt) stage x-tiles in the recurrent kernels.
// ---------------------------------------------------------------------------

typedef __attribute__((ext_vector_type(16))) _Float16 v16h;
typedef __attribute__((ext_vector_type(8)))  float    v8f;

#define S_   400
#define B_   32
#define T_   64
#define E_   128
#define H_   256
#define V_   50000
#define OOV_ 50
#define VX_  50050
#define H2_  512
#define H4_  1024
#define BOS_ 2

__device__ __forceinline__ float sigf(float x) { return 1.f / (1.f + __expf(-x)); }

// Opaque copy of a uniform pointer: blocks LICM from hoisting (and then
// spilling) loop-invariant weight-fragment loads out of the time loop.
#define LAUNDER_S(p) asm volatile("" : "+s"(p))

// ---- CDNA5 async global->LDS copy (tracked by ASYNCcnt) -------------------
// Generic LDS addresses carry the LDS byte offset in the low 32 bits.
__device__ __forceinline__ void async_g2l_b64(const void* gptr, void* lptr) {
  unsigned lds_off = (unsigned)(unsigned long long)lptr;
  asm volatile("global_load_async_to_lds_b64 %0, %1, off"
               :: "v"(lds_off), "v"(gptr) : "memory");
}
__device__ __forceinline__ void wait_async0() {
  asm volatile("s_wait_asynccnt 0x0" ::: "memory");
}

// ---- WMMA fragment helpers (ISA 7.12.2 layouts, wave32) -------------------
// A (16x32 f16, MxK): lane L holds row m=L&15.
//   L<16 : halves 0..7 -> k=0..7,  halves 8..15 -> k=16..23
//   L>=16: halves 0..7 -> k=8..15, halves 8..15 -> k=24..31
__device__ __forceinline__ v16h load_a(const _Float16* A, int lda, int m0, int kb) {
  int lane = threadIdx.x & 31;
  const _Float16* p = A + (m0 + (lane & 15)) * lda + kb + ((lane >> 4) << 3);
  v16h f;
#pragma unroll
  for (int h = 0; h < 8; ++h) { f[h] = p[h]; f[h + 8] = p[h + 16]; }
  return f;
}
__device__ __forceinline__ v16h load_a_f32(const float* A, int lda, int m0, int kb) {
  int lane = threadIdx.x & 31;
  const float* p = A + (m0 + (lane & 15)) * lda + kb + ((lane >> 4) << 3);
  v16h f;
#pragma unroll
  for (int h = 0; h < 8; ++h) { f[h] = (_Float16)p[h]; f[h + 8] = (_Float16)p[h + 16]; }
  return f;
}
// B (32x16 f16, KxN), sourced from row-major W[N,K] (B(k,n)=W[n,k]):
// lane L holds col n=L&15; same k order as A (L<16: k=0..7,16..23; L>=16: k=8..15,24..31).
__device__ __forceinline__ v16h load_b(const _Float16* W, int ldw, int n0, int kb) {
  int lane = threadIdx.x & 31;
  const _Float16* p = W + (n0 + (lane & 15)) * ldw + kb + ((lane >> 4) << 3);
  v16h f;
#pragma unroll
  for (int h = 0; h < 8; ++h) { f[h] = p[h]; f[h + 8] = p[h + 16]; }
  return f;
}
__device__ __forceinline__ v8f wmma16(v16h a, v16h b, v8f c) {
  return __builtin_amdgcn_wmma_f32_16x16x32_f16(false, a, false, b, (short)0, c, false, false);
}
// C/D (16x16 f32): lane n = lane&15, VGPR i -> row m = i + 8*(lane>=16).

// ---- small utility kernels -------------------------------------------------
__global__ void cvt_kernel(const float* __restrict__ s, _Float16* __restrict__ d, int n) {
  int i = blockIdx.x * blockDim.x + threadIdx.x;
  if (i < n) d[i] = (_Float16)s[i];
}

__global__ void embed_kernel(const int* __restrict__ inputs, const _Float16* __restrict__ emb,
                             _Float16* __restrict__ out, int total) {
  int i = blockIdx.x * blockDim.x + threadIdx.x;
  if (i < total) {
    int r = i >> 7, e = i & 127;
    out[i] = emb[(long)inputs[r] * E_ + e];
  }
}

// ---- persistent bidirectional-encoder LSTM (1 workgroup per direction) ----
__global__ __launch_bounds__(1024) void encoder_kernel(
    const _Float16* __restrict__ emb16,
    const _Float16* __restrict__ WihF, const _Float16* __restrict__ WhhF,
    const float* __restrict__ bihF, const float* __restrict__ bhhF,
    const _Float16* __restrict__ WihB, const _Float16* __restrict__ WhhB,
    const float* __restrict__ bihB, const float* __restrict__ bhhB,
    float* __restrict__ enc_out, _Float16* __restrict__ enc_out16,
    float* __restrict__ hF, float* __restrict__ cF,
    float* __restrict__ hB, float* __restrict__ cB) {
  __shared__ _Float16 hl[B_ * H_];          // 16KB shared hidden state
  __shared__ _Float16 xst[2][B_ * E_];      // 2 x 8KB async-staged x tiles
  int dir = blockIdx.x;
  const _Float16* Wih = dir ? WihB : WihF;
  const _Float16* Whh = dir ? WhhB : WhhF;
  const float* bih = dir ? bihB : bihF;
  const float* bhh = dir ? bhhB : bhhF;
  float* hfin = dir ? hB : hF;
  float* cfin = dir ? cB : cF;

  int tid = threadIdx.x, w = tid >> 5, lane = tid & 31;
  int l15 = lane & 15, lhi = (lane >> 4) << 3;
  int mt = w >> 4, jt = w & 15;             // wave owns (m-tile, hidden slice)
  int jt16 = jt * 16;
  for (int i = tid; i < B_ * H_; i += 1024) hl[i] = (_Float16)0.f;
  v8f creg = {};                            // cell state lives in registers

  // stage x tile for t=0 (ASYNCcnt path)
  {
    int s0 = dir ? (S_ - 1) : 0;
    async_g2l_b64(emb16 + (long)s0 * B_ * E_ + tid * 4, &xst[0][tid * 4]);
    wait_async0();
  }
  __syncthreads();

  for (int t = 0; t < S_; ++t) {
    int cur = t & 1;
    int s = dir ? (S_ - 1 - t) : t;
    // prefetch next x tile into the other buffer while we compute
    if (t + 1 < S_) {
      int s1 = dir ? (S_ - 1 - (t + 1)) : (t + 1);
      async_g2l_b64(emb16 + (long)s1 * B_ * E_ + tid * 4, &xst[1 - cur][tid * 4]);
    }
    const _Float16* X = xst[cur];

    // Re-materialize weight pointers each step so LICM cannot hoist (and
    // spill) the 48 weight fragments; they stay cheap per-step L2 loads.
    const _Float16* WihL = Wih;
    const _Float16* WhhL = Whh;
    LAUNDER_S(WihL);
    LAUNDER_S(WhhL);

    v8f g0 = {}, g1 = {}, g2 = {}, g3 = {};
#pragma unroll 1
    for (int kb = 0; kb < E_; kb += 32) {
      v16h a = load_a(X, E_, mt * 16, kb);
      g0 = wmma16(a, load_b(WihL, E_, jt16, kb), g0);
      g1 = wmma16(a, load_b(WihL, E_, H_ + jt16, kb), g1);
      g2 = wmma16(a, load_b(WihL, E_, 2 * H_ + jt16, kb), g2);
      g3 = wmma16(a, load_b(WihL, E_, 3 * H_ + jt16, kb), g3);
    }
#pragma unroll 1
    for (int kb = 0; kb < H_; kb += 32) {
      v16h a = load_a(hl, H_, mt * 16, kb);
      g0 = wmma16(a, load_b(WhhL, H_, jt16, kb), g0);
      g1 = wmma16(a, load_b(WhhL, H_, H_ + jt16, kb), g1);
      g2 = wmma16(a, load_b(WhhL, H_, 2 * H_ + jt16, kb), g2);
      g3 = wmma16(a, load_b(WhhL, H_, 3 * H_ + jt16, kb), g3);
    }
    __syncthreads();                         // all reads of h done
    int n = jt16 + l15;
    float b0 = bih[n] + bhh[n];
    float b1 = bih[H_ + n] + bhh[H_ + n];
    float b2 = bih[2 * H_ + n] + bhh[2 * H_ + n];
    float b3 = bih[3 * H_ + n] + bhh[3 * H_ + n];
#pragma unroll
    for (int i = 0; i < 8; ++i) {
      int m = mt * 16 + lhi + i;
      float ig = sigf(g0[i] + b0), fg = sigf(g1[i] + b1);
      float gg = tanhf(g2[i] + b2), og = sigf(g3[i] + b3);
      float cn = fg * creg[i] + ig * gg;
      creg[i] = cn;
      float hn = og * tanhf(cn);
      hl[m * H_ + n] = (_Float16)hn;
      long off = ((long)s * B_ + m) * H2_ + dir * H_ + n;
      enc_out[off] = hn;
      enc_out16[off] = (_Float16)hn;
    }
    wait_async0();                           // next x tile landed in LDS
    __syncthreads();                         // h (and x) ready for next step
  }
  {
    int n = jt16 + l15;
#pragma unroll
    for (int i = 0; i < 8; ++i) cfin[(mt * 16 + lhi + i) * H_ + n] = creg[i];
  }
  for (int i = tid; i < B_ * H_; i += 1024) hfin[i] = (float)hl[i];
}

// ---- enc_feat = enc_out @ feat_W.T : (12800 x 512 x 512) WMMA GEMM --------
__global__ __launch_bounds__(256) void encfeat_kernel(const _Float16* __restrict__ A,
                                                      const _Float16* __restrict__ W,
                                                      float* __restrict__ C) {
  int gw = (blockIdx.x * 256 + threadIdx.x) >> 5;  // exact grid: 800 x 32 tiles
  int mt = gw >> 5, nt = gw & 31;
  v8f acc = {};
#pragma unroll 1
  for (int kb = 0; kb < H2_; kb += 32)
    acc = wmma16(load_a(A, H2_, mt * 16, kb), load_b(W, H2_, nt * 16, kb), acc);
  int lane = threadIdx.x & 31, l15 = lane & 15, lhi = (lane >> 4) << 3;
  int n = nt * 16 + l15;
#pragma unroll
  for (int i = 0; i < 8; ++i) C[((long)(mt * 16 + lhi + i)) * H2_ + n] = acc[i];
}

// ---- h_dec/c_dec = relu(concat(f,b) @ red_W.T + b) ------------------------
__global__ __launch_bounds__(1024) void reduce_kernel(
    const float* __restrict__ hF, const float* __restrict__ hB,
    const float* __restrict__ cF, const float* __restrict__ cB,
    const _Float16* __restrict__ WH, const float* __restrict__ bH,
    const _Float16* __restrict__ WC, const float* __restrict__ bC,
    float* __restrict__ hdec, float* __restrict__ cdec) {
  const float* A0 = blockIdx.x ? cF : hF;
  const float* A1 = blockIdx.x ? cB : hB;
  const _Float16* W = blockIdx.x ? WC : WH;
  const float* bb = blockIdx.x ? bC : bH;
  float* out = blockIdx.x ? cdec : hdec;
  int tid = threadIdx.x, w = tid >> 5, lane = tid & 31;
  int l15 = lane & 15, lhi = (lane >> 4) << 3;
  int mt = w >> 4, nt = w & 15;
  v8f acc = {};
#pragma unroll 1
  for (int kb = 0; kb < H_; kb += 32)
    acc = wmma16(load_a_f32(A0, H_, mt * 16, kb), load_b(W, H2_, nt * 16, kb), acc);
#pragma unroll 1
  for (int kb = 0; kb < H_; kb += 32)
    acc = wmma16(load_a_f32(A1, H_, mt * 16, kb), load_b(W, H2_, nt * 16, H_ + kb), acc);
  int n = nt * 16 + l15;
  float bn = bb[n];
#pragma unroll
  for (int i = 0; i < 8; ++i) {
    float v = acc[i] + bn;
    out[(mt * 16 + lhi + i) * H_ + n] = v > 0.f ? v : 0.f;
  }
}

// ---- persistent decoder: LSTM + attention + p_gen + out1 (1 workgroup) ----
__global__ __launch_bounds__(1024) void decoder_kernel(
    const int* __restrict__ inputs, const int* __restrict__ targets,
    const _Float16* __restrict__ embt16,
    const _Float16* __restrict__ ctxW, const float* __restrict__ ctxb,
    const _Float16* __restrict__ dWih, const _Float16* __restrict__ dWhh,
    const float* __restrict__ dbih, const float* __restrict__ dbhh,
    const _Float16* __restrict__ afW, const float* __restrict__ afb,
    const float* __restrict__ attn_v,
    const float* __restrict__ pgenW, const float* __restrict__ pgenb,
    const _Float16* __restrict__ out1W, const float* __restrict__ out1b,
    const float* __restrict__ enc_feat, const float* __restrict__ enc_out,
    const float* __restrict__ hdec, const float* __restrict__ cdec,
    _Float16* __restrict__ hid16, float* __restrict__ pgen_out,
    float* __restrict__ attns_out) {
  // ~197KB static LDS: legal on CDNA5 (320KB / WGP)
  __shared__ _Float16 h16[B_ * H_];    // 16KB
  __shared__ _Float16 c16[B_ * H_];    // 16KB
  __shared__ _Float16 ctx16[B_ * H2_]; // 32KB (ctx in / context out, reused)
  __shared__ _Float16 x16[B_ * E_];    // 8KB (async-gathered embeddings)
  __shared__ _Float16 inp16[B_ * E_];  // 8KB
  __shared__ float dfeat[B_ * H2_];    // 64KB
  __shared__ float sc[S_ * B_];        // 51.2KB attention scores
  __shared__ float avl[H2_];

  int tid = threadIdx.x, w = tid >> 5, lane = tid & 31;
  int l15 = lane & 15, lhi = (lane >> 4) << 3;
  int mtw = w >> 4, jtw = w & 15;
  int jt16 = jtw * 16;

  for (int i = tid; i < B_ * H_; i += 1024) h16[i] = (_Float16)hdec[i];
  for (int i = tid; i < B_ * H2_; i += 1024) ctx16[i] = (_Float16)0.f;
  if (tid < H2_) avl[tid] = attn_v[tid];
  v8f creg;
  {
    int n = jt16 + l15;
#pragma unroll
    for (int i = 0; i < 8; ++i) creg[i] = cdec[(mtw * 16 + lhi + i) * H_ + n];
  }
  __syncthreads();

  for (int t = 0; t < T_; ++t) {
    // per-step opaque weight pointers: keep fragment loads inside the loop
    const _Float16* ctxWL = ctxW;  LAUNDER_S(ctxWL);
    const _Float16* dWihL = dWih;  LAUNDER_S(dWihL);
    const _Float16* dWhhL = dWhh;  LAUNDER_S(dWhhL);
    const _Float16* afWL  = afW;   LAUNDER_S(afWL);
    const _Float16* out1WL = out1W; LAUNDER_S(out1WL);

    // x = embedding[tok]: per-lane async gather straight into LDS
    {
      int b = tid >> 5, e0 = (tid & 31) * 4;
      int tok = (t == 0) ? BOS_ : targets[(t - 1) * B_ + b];
      async_g2l_b64(embt16 + (long)tok * E_ + e0, &x16[b * E_ + e0]);
      wait_async0();
    }
    __syncthreads();

    // inp = concat(ctx, x) @ ctx_W.T + ctx_b  (16 tiles -> waves 0..15)
    if (w < 16) {
      int mt = w >> 3, nt = w & 7;
      v8f acc = {};
#pragma unroll 1
      for (int kb = 0; kb < H2_; kb += 32)
        acc = wmma16(load_a(ctx16, H2_, mt * 16, kb), load_b(ctxWL, H2_ + E_, nt * 16, kb), acc);
#pragma unroll 1
      for (int kb = 0; kb < E_; kb += 32)
        acc = wmma16(load_a(x16, E_, mt * 16, kb), load_b(ctxWL, H2_ + E_, nt * 16, H2_ + kb), acc);
      int n = nt * 16 + l15;
      float bb = ctxb[n];
#pragma unroll
      for (int i = 0; i < 8; ++i) inp16[(mt * 16 + lhi + i) * E_ + n] = (_Float16)(acc[i] + bb);
    }
    __syncthreads();

    // LSTM gates: all 32 waves, k-outer / 4-gate-inner
    v8f g0 = {}, g1 = {}, g2 = {}, g3 = {};
#pragma unroll 1
    for (int kb = 0; kb < E_; kb += 32) {
      v16h a = load_a(inp16, E_, mtw * 16, kb);
      g0 = wmma16(a, load_b(dWihL, E_, jt16, kb), g0);
      g1 = wmma16(a, load_b(dWihL, E_, H_ + jt16, kb), g1);
      g2 = wmma16(a, load_b(dWihL, E_, 2 * H_ + jt16, kb), g2);
      g3 = wmma16(a, load_b(dWihL, E_, 3 * H_ + jt16, kb), g3);
    }
#pragma unroll 1
    for (int kb = 0; kb < H_; kb += 32) {
      v16h a = load_a(h16, H_, mtw * 16, kb);
      g0 = wmma16(a, load_b(dWhhL, H_, jt16, kb), g0);
      g1 = wmma16(a, load_b(dWhhL, H_, H_ + jt16, kb), g1);
      g2 = wmma16(a, load_b(dWhhL, H_, 2 * H_ + jt16, kb), g2);
      g3 = wmma16(a, load_b(dWhhL, H_, 3 * H_ + jt16, kb), g3);
    }
    __syncthreads();
    {
      int n = jt16 + l15;
      float b0 = dbih[n] + dbhh[n];
      float b1 = dbih[H_ + n] + dbhh[H_ + n];
      float b2 = dbih[2 * H_ + n] + dbhh[2 * H_ + n];
      float b3 = dbih[3 * H_ + n] + dbhh[3 * H_ + n];
#pragma unroll
      for (int i = 0; i < 8; ++i) {
        int m = mtw * 16 + lhi + i;
        float ig = sigf(g0[i] + b0), fg = sigf(g1[i] + b1);
        float gg = tanhf(g2[i] + b2), og = sigf(g3[i] + b3);
        float cn = fg * creg[i] + ig * gg;
        creg[i] = cn;
        float hn = og * tanhf(cn);
        h16[m * H_ + n] = (_Float16)hn;
        c16[m * H_ + n] = (_Float16)cn;
      }
    }
    __syncthreads();

    // dec_feat = concat(h,c) @ attn_feat_W.T + b  (64 tiles, 2/wave)
#pragma unroll
    for (int rep = 0; rep < 2; ++rep) {
      int tix = w + rep * 32, mt = tix >> 5, nt = tix & 31;
      v8f acc = {};
#pragma unroll 1
      for (int kb = 0; kb < H_; kb += 32)
        acc = wmma16(load_a(h16, H_, mt * 16, kb), load_b(afWL, H2_, nt * 16, kb), acc);
#pragma unroll 1
      for (int kb = 0; kb < H_; kb += 32)
        acc = wmma16(load_a(c16, H_, mt * 16, kb), load_b(afWL, H2_, nt * 16, H_ + kb), acc);
      int n = nt * 16 + l15;
      float bb = afb[n];
#pragma unroll
      for (int i = 0; i < 8; ++i) dfeat[(mt * 16 + lhi + i) * H2_ + n] = acc[i] + bb;
    }
    __syncthreads();

    // scores[s,b] = tanh(enc_feat + dec_feat) . attn_v
    for (int r = tid; r < S_ * B_; r += 1024) {
      const float* ef = enc_feat + (long)r * H2_;
      const float* df = dfeat + (r & 31) * H2_;
      float a0 = 0.f;
      for (int d = 0; d < H2_; ++d) a0 += avl[d] * tanhf(ef[d] + df[d]);
      sc[r] = a0;
    }
    __syncthreads();

    // masked softmax over s, per batch column
    if (tid < B_) {
      int b = tid;
      float mx = -3.4e38f;
      for (int s = 0; s < S_; ++s) mx = fmaxf(mx, sc[s * B_ + b]);
      float sum = 0.f;
      for (int s = 0; s < S_; ++s) {
        float e = (inputs[s * B_ + b] > 0) ? __expf(sc[s * B_ + b] - mx) : 0.f;
        sc[s * B_ + b] = e;
        sum += e;
      }
      float inv = 1.f / sum;
      for (int s = 0; s < S_; ++s) {
        float a = sc[s * B_ + b] * inv;
        sc[s * B_ + b] = a;
        attns_out[((long)t * S_ + s) * B_ + b] = a;
      }
    }
    __syncthreads();

    // context[b,d] = sum_s a[s,b] * enc_out[s,b,d]   (reuse ctx16)
    for (int idx = tid; idx < B_ * H2_; idx += 1024) {
      int b = idx >> 9, d = idx & (H2_ - 1);
      float a0 = 0.f;
      for (int s = 0; s < S_; ++s) a0 += sc[s * B_ + b] * enc_out[((long)s * B_ + b) * H2_ + d];
      ctx16[idx] = (_Float16)a0;
    }
    __syncthreads();

    // p_gen = sigmoid([context, h, c, inp] . pgen_W + b)
    if (tid < B_) {
      int b = tid;
      float a0 = pgenb[0];
      for (int d = 0; d < H2_; ++d) a0 += pgenW[d] * (float)ctx16[b * H2_ + d];
      for (int d = 0; d < H_; ++d) a0 += pgenW[H2_ + d] * (float)h16[b * H_ + d];
      for (int d = 0; d < H_; ++d) a0 += pgenW[H2_ + H_ + d] * (float)c16[b * H_ + d];
      for (int d = 0; d < E_; ++d) a0 += pgenW[H4_ + d] * (float)inp16[b * E_ + d];
      pgen_out[t * B_ + b] = sigf(a0);
    }

    // hid = concat(h, c, context) @ out1_W.T + b  (32 tiles, 1/wave)
    {
      int mt = mtw, nt = jtw;
      v8f acc = {};
#pragma unroll 1
      for (int kb = 0; kb < H_; kb += 32)
        acc = wmma16(load_a(h16, H_, mt * 16, kb), load_b(out1WL, H4_, nt * 16, kb), acc);
#pragma unroll 1
      for (int kb = 0; kb < H_; kb += 32)
        acc = wmma16(load_a(c16, H_, mt * 16, kb), load_b(out1WL, H4_, nt * 16, H_ + kb), acc);
#pragma unroll 1
      for (int kb = 0; kb < H2_; kb += 32)
        acc = wmma16(load_a(ctx16, H2_, mt * 16, kb), load_b(out1WL, H4_, nt * 16, H2_ + kb), acc);
      int n = nt * 16 + l15;
      float bb = out1b[n];
#pragma unroll
      for (int i = 0; i < 8; ++i)
        hid16[((long)t * B_ + mt * 16 + lhi + i) * H_ + n] = (_Float16)(acc[i] + bb);
    }
    __syncthreads();
  }
}

// ---- vocab logits: (2048 x 256 x 50000) WMMA GEMM, fully parallel ---------
__global__ __launch_bounds__(256) void logits_kernel(const _Float16* __restrict__ hid,
                                                     const _Float16* __restrict__ W2,
                                                     const float* __restrict__ b2,
                                                     float* __restrict__ dists) {
  long gw = ((long)blockIdx.x * 256 + threadIdx.x) >> 5; // exact: 128 x 3125 tiles
  int nt = (int)(gw % 3125);
  int mt = (int)(gw / 3125);
  __builtin_prefetch(W2 + ((long)nt * 16 + 16) * H_, 0, 0); // global_prefetch_b8
  v8f acc = {};
#pragma unroll 1
  for (int kb = 0; kb < H_; kb += 32)
    acc = wmma16(load_a(hid, H_, mt * 16, kb), load_b(W2, H_, nt * 16, kb), acc);
  int lane = threadIdx.x & 31, l15 = lane & 15, lhi = (lane >> 4) << 3;
  int n = nt * 16 + l15;
  float bn = b2[n];
#pragma unroll
  for (int i = 0; i < 8; ++i) dists[(long)(mt * 16 + lhi + i) * VX_ + n] = acc[i] + bn;
}

// ---- per-row softmax * p_gen, zero OOV tail --------------------------------
__global__ __launch_bounds__(256) void vsoftmax_kernel(float* __restrict__ dists,
                                                       const float* __restrict__ pgen) {
  __shared__ float sred[256];
  int r = blockIdx.x, tid = threadIdx.x;
  float* row = dists + (long)r * VX_;
  float mx = -3.4e38f;
  for (int v = tid; v < V_; v += 256) mx = fmaxf(mx, row[v]);
  sred[tid] = mx;
  __syncthreads();
  for (int s2 = 128; s2 > 0; s2 >>= 1) {
    if (tid < s2) sred[tid] = fmaxf(sred[tid], sred[tid + s2]);
    __syncthreads();
  }
  mx = sred[0];
  __syncthreads();
  float sum = 0.f;
  for (int v = tid; v < V_; v += 256) sum += __expf(row[v] - mx);
  sred[tid] = sum;
  __syncthreads();
  for (int s2 = 128; s2 > 0; s2 >>= 1) {
    if (tid < s2) sred[tid] += sred[tid + s2];
    __syncthreads();
  }
  float scale = pgen[r] / sred[0];
  for (int v = tid; v < V_; v += 256) row[v] = __expf(row[v] - mx) * scale;
  for (int v = V_ + tid; v < VX_; v += 256) row[v] = 0.f;
}

// ---- copy-distribution scatter-add ----------------------------------------
__global__ void scatter_kernel(const int* __restrict__ iext, const float* __restrict__ pgen,
                               const float* __restrict__ attns, float* __restrict__ dists) {
  long i = (long)blockIdx.x * blockDim.x + threadIdx.x;
  if (i >= (long)T_ * B_ * S_) return;
  int s = (int)(i % S_);
  long tb = i / S_;
  int b = (int)(tb % B_);
  int t = (int)(tb / B_);
  float a = attns[((long)t * S_ + s) * B_ + b];
  float pg = pgen[t * B_ + b];
  int idx = iext[s * B_ + b];
  if (idx >= 0 && idx < VX_)
    atomicAdd(&dists[((long)t * B_ + b) * VX_ + idx], (1.f - pg) * a);
}

// ---- per-row argmax -> tokens ---------------------------------------------
__global__ __launch_bounds__(256) void argmax_kernel(const float* __restrict__ dists,
                                                     float* __restrict__ tokens) {
  __shared__ float sv[256];
  __shared__ int si[256];
  int r = blockIdx.x, tid = threadIdx.x;
  const float* row = dists + (long)r * VX_;
  float bv = -3.4e38f;
  int bi = 0;
  for (int v = tid; v < VX_; v += 256) {
    float x = row[v];
    if (x > bv) { bv = x; bi = v; }
  }
  sv[tid] = bv;
  si[tid] = bi;
  __syncthreads();
  for (int s2 = 128; s2 > 0; s2 >>= 1) {
    if (tid < s2) {
      if (sv[tid + s2] > sv[tid] || (sv[tid + s2] == sv[tid] && si[tid + s2] < si[tid])) {
        sv[tid] = sv[tid + s2];
        si[tid] = si[tid + s2];
      }
    }
    __syncthreads();
  }
  if (tid == 0) tokens[r] = (float)si[0];
}

// ---------------------------------------------------------------------------
extern "C" void kernel_launch(void* const* d_in, const int* in_sizes, int n_in,
                              void* d_out, int out_size, void* d_ws, size_t ws_size,
                              hipStream_t stream) {
  const int* inputs = (const int*)d_in[0];
  const int* iext = (const int*)d_in[2];
  const int* targets = (const int*)d_in[3];
  const float* embedding = (const float*)d_in[5];
  const float* WihF = (const float*)d_in[6], * WhhF = (const float*)d_in[7];
  const float* bihF = (const float*)d_in[8], * bhhF = (const float*)d_in[9];
  const float* WihB = (const float*)d_in[10], * WhhB = (const float*)d_in[11];
  const float* bihB = (const float*)d_in[12], * bhhB = (const float*)d_in[13];
  const float* featW = (const float*)d_in[14];
  const float* redhW = (const float*)d_in[15], * redhb = (const float*)d_in[16];
  const float* redcW = (const float*)d_in[17], * redcb = (const float*)d_in[18];
  const float* ctxW = (const float*)d_in[19], * ctxb = (const float*)d_in[20];
  const float* dWih = (const float*)d_in[21], * dWhh = (const float*)d_in[22];
  const float* dbih = (const float*)d_in[23], * dbhh = (const float*)d_in[24];
  const float* afW = (const float*)d_in[25], * afb = (const float*)d_in[26];
  const float* attnv = (const float*)d_in[27];
  const float* pgenW = (const float*)d_in[28], * pgenb = (const float*)d_in[29];
  const float* out1W = (const float*)d_in[30], * out1b = (const float*)d_in[31];
  const float* out2W = (const float*)d_in[32], * out2b = (const float*)d_in[33];

  float* dists = (float*)d_out;                       // [T,B,V+OOV]
  float* tokens = dists + (long)T_ * B_ * VX_;        // [T,B]
  float* attns = tokens + T_ * B_;                    // [T,S,B]

  char* ws = (char*)d_ws;
  size_t off = 0;
  auto alloc = [&](size_t bytes) -> void* {
    void* p = ws + off;
    off = (off + bytes + 255) & ~(size_t)255;
    return p;
  };

  _Float16* embt16 = (_Float16*)alloc((size_t)V_ * E_ * 2);   // fp16 embed table
  _Float16* emb16 = (_Float16*)alloc((size_t)S_ * B_ * E_ * 2);
  _Float16* WihF16 = (_Float16*)alloc((size_t)H4_ * E_ * 2);
  _Float16* WhhF16 = (_Float16*)alloc((size_t)H4_ * H_ * 2);
  _Float16* WihB16 = (_Float16*)alloc((size_t)H4_ * E_ * 2);
  _Float16* WhhB16 = (_Float16*)alloc((size_t)H4_ * H_ * 2);
  _Float16* featW16 = (_Float16*)alloc((size_t)H2_ * H2_ * 2);
  _Float16* redhW16 = (_Float16*)alloc((size_t)H_ * H2_ * 2);
  _Float16* redcW16 = (_Float16*)alloc((size_t)H_ * H2_ * 2);
  _Float16* ctxW16 = (_Float16*)alloc((size_t)E_ * (H2_ + E_) * 2);
  _Float16* dWih16 = (_Float16*)alloc((size_t)H4_ * E_ * 2);
  _Float16* dWhh16 = (_Float16*)alloc((size_t)H4_ * H_ * 2);
  _Float16* afW16 = (_Float16*)alloc((size_t)H2_ * H2_ * 2);
  _Float16* out1W16 = (_Float16*)alloc((size_t)H_ * H4_ * 2);
  _Float16* out2W16 = (_Float16*)alloc((size_t)V_ * H_ * 2);
  float* enc_out = (float*)alloc((size_t)S_ * B_ * H2_ * 4);
  _Float16* enc_out16 = (_Float16*)alloc((size_t)S_ * B_ * H2_ * 2);
  float* enc_feat = (float*)alloc((size_t)S_ * B_ * H2_ * 4);
  float* hF = (float*)alloc((size_t)B_ * H_ * 4);
  float* cF = (float*)alloc((size_t)B_ * H_ * 4);
  float* hB = (float*)alloc((size_t)B_ * H_ * 4);
  float* cB = (float*)alloc((size_t)B_ * H_ * 4);
  float* hdec = (float*)alloc((size_t)B_ * H_ * 4);
  float* cdec = (float*)alloc((size_t)B_ * H_ * 4);
  _Float16* hid16 = (_Float16*)alloc((size_t)T_ * B_ * H_ * 2);
  float* pgen = (float*)alloc((size_t)T_ * B_ * 4);

  auto cvt = [&](const float* src, _Float16* dst, int n) {
    cvt_kernel<<<(n + 255) / 256, 256, 0, stream>>>(src, dst, n);
  };
  cvt(embedding, embt16, V_ * E_);
  cvt(WihF, WihF16, H4_ * E_);
  cvt(WhhF, WhhF16, H4_ * H_);
  cvt(WihB, WihB16, H4_ * E_);
  cvt(WhhB, WhhB16, H4_ * H_);
  cvt(featW, featW16, H2_ * H2_);
  cvt(redhW, redhW16, H_ * H2_);
  cvt(redcW, redcW16, H_ * H2_);
  cvt(ctxW, ctxW16, E_ * (H2_ + E_));
  cvt(dWih, dWih16, H4_ * E_);
  cvt(dWhh, dWhh16, H4_ * H_);
  cvt(afW, afW16, H2_ * H2_);
  cvt(out1W, out1W16, H_ * H4_);
  cvt(out2W, out2W16, V_ * H_);

  embed_kernel<<<(S_ * B_ * E_ + 255) / 256, 256, 0, stream>>>(inputs, embt16, emb16,
                                                               S_ * B_ * E_);

  encoder_kernel<<<2, 1024, 0, stream>>>(emb16, WihF16, WhhF16, bihF, bhhF, WihB16, WhhB16,
                                         bihB, bhhB, enc_out, enc_out16, hF, cF, hB, cB);

  encfeat_kernel<<<3200, 256, 0, stream>>>(enc_out16, featW16, enc_feat);

  reduce_kernel<<<2, 1024, 0, stream>>>(hF, hB, cF, cB, redhW16, redhb, redcW16, redcb,
                                        hdec, cdec);

  decoder_kernel<<<1, 1024, 0, stream>>>(inputs, targets, embt16, ctxW16, ctxb, dWih16,
                                         dWhh16, dbih, dbhh, afW16, afb, attnv, pgenW, pgenb,
                                         out1W16, out1b, enc_feat, enc_out, hdec, cdec, hid16,
                                         pgen, attns);

  logits_kernel<<<50000, 256, 0, stream>>>(hid16, out2W16, out2b, dists);

  vsoftmax_kernel<<<T_ * B_, 256, 0, stream>>>(dists, pgen);

  scatter_kernel<<<(T_ * B_ * S_ + 255) / 256, 256, 0, stream>>>(iext, pgen, attns, dists);

  argmax_kernel<<<T_ * B_, 256, 0, stream>>>(dists, tokens);

  (void)in_sizes; (void)n_in; (void)out_size; (void)ws_size;
}
